// SPDGMultiHeadAttention_22436909154792
// MI455X (gfx1250) — matrix operand order repaired and checked by reading, not compiled, not run on hardware
//
#include <hip/hip_runtime.h>
#include <hip/hip_bf16.h>
#include <math.h>

// ---------------------------------------------------------------------------
// CDNA5 / gfx1250 implementation. wave32.
//  - All GEMMs via v_wmma_f32_16x16x32_bf16 (f32 accumulate).
//  - GEMM staging via global_load_async_to_lds_b128 (ASYNCcnt) double-buffered.
//  - Flash attention softmax reductions via DPP16 (no LDS round trips).
// ---------------------------------------------------------------------------

typedef __hip_bfloat16 bf16_t;
typedef __bf16 v16bf __attribute__((ext_vector_type(16)));
typedef float  v8f   __attribute__((ext_vector_type(8)));

#define DEV __device__ __forceinline__

constexpr int   Bb   = 2;
constexpr int   Ss   = 1024;
constexpr int   Dd   = 512;
constexpr int   Hh   = 8;
constexpr int   DHd  = 64;
constexpr int   Ff   = 2048;     // 4*D
constexpr int   WHalf = 51;      // w=102 -> window [q-51, q+51]
constexpr float NEGV = -1.0e9f;

union FragU { uint4 q[2]; v16bf v; };

// A fragment (16x32 bf16, M x K): lane = M row (both halves), K runs
// [kb,kb+8) and [kb+16,kb+24) with kb = 0 (lanes 0-15) or 8 (lanes 16-31).
DEV v16bf frag_a_load(const bf16_t* tile, int ld, int lane) {
  int row = lane & 15;
  int kb  = (lane & 16) ? 8 : 0;
  const bf16_t* p = tile + (size_t)row * ld + kb;
  FragU f;
  f.q[0] = *(const uint4*)(p);
  f.q[1] = *(const uint4*)(p + 16);
  return f.v;
}

// B fragment (32x16 bf16, K x N) sourced from a row-major [N,K] tile:
// lane = N column, K runs [kb,kb+8) and [kb+16,kb+24) with kb = 0 / 8 per half-wave.
DEV v16bf frag_b_load(const bf16_t* tile, int ld, int lane) {
  int row = lane & 15;
  int kb  = (lane & 16) ? 8 : 0;
  const bf16_t* p = tile + (size_t)row * ld + kb;
  FragU f;
  f.q[0] = *(const uint4*)(p);
  f.q[1] = *(const uint4*)(p + 16);
  return f.v;
}

DEV v8f wmma_bf16(v16bf a, v16bf b, v8f c) {
  return __builtin_amdgcn_wmma_f32_16x16x32_bf16(
      /*neg_a=*/false, a, /*neg_b=*/false, b,
      /*c_mod=*/(short)0, c, /*reuse_a=*/false, /*reuse_b=*/false);
}

// Async global -> LDS copy, 16B per lane (GLOBAL_LOAD_ASYNC_TO_LDS_B128, GV mode).
// Low 32 bits of the generic shared pointer are the LDS byte offset
// (flat->LDS addresses truncate to addr[31:0] per the CDNA5 ISA).
DEV void async_copy_b128(const bf16_t* gptr, bf16_t* lptr) {
  unsigned long long ga = (unsigned long long)(uintptr_t)gptr;
  unsigned la = (unsigned)(uintptr_t)lptr;
  asm volatile("global_load_async_to_lds_b128 %0, %1, off"
               :: "v"(la), "v"(ga) : "memory");
}

// DPP16 butterfly reduction across a 16-lane half (wave32 halves hold
// distinct C-matrix rows). Pure VALU: no LDS traffic, no dscnt waits.
template<int CTRL>
DEV float dppf(float x) {
  return __builtin_bit_cast(float,
      __builtin_amdgcn_mov_dpp(__builtin_bit_cast(int, x), CTRL, 0xf, 0xf, true));
}
DEV float rowmax16(float v) {
  v = fmaxf(v, dppf<0xB1>(v));    // quad_perm(1,0,3,2)  : xor 1
  v = fmaxf(v, dppf<0x4E>(v));    // quad_perm(2,3,0,1)  : xor 2
  v = fmaxf(v, dppf<0x141>(v));   // row_half_mirror     : combine quads in 8
  v = fmaxf(v, dppf<0x140>(v));   // row_mirror          : combine 8-groups
  return v;
}
DEV float rowsum16(float v) {
  v += dppf<0xB1>(v);
  v += dppf<0x4E>(v);
  v += dppf<0x141>(v);
  v += dppf<0x140>(v);
  return v;
}

// ---------------------------------------------------------------------------
// Generic bf16 GEMM: C[M,N] = A[M,K] @ W[N,K]^T, fused epilogues.
// Block: 256 threads (8 waves), tile 128(M) x 64(N), K-step 32.
// Double-buffered LDS fed by async global->LDS copies: tile t+1's fetch is
// fully overlapped with tile t's WMMA work (s_wait_asynccnt 3 = previous
// tile's 3 in-order async ops complete).
// EPI 0: outB = bf16((acc + bias[n]) * scaleOut)                  (Q/K proj)
// EPI 1: outB = bf16(acc + bias[n]) stored transposed [B,H,DH,S]  (V proj)
// EPI 2: outF = acc + bias[n] + res[m,n]                          (residual)
// EPI 3: outB = bf16(gelu(acc + bias[n]))                         (FFN1)
// ---------------------------------------------------------------------------
template<int EPI>
__global__ __launch_bounds__(256)
void gemm_bf16_kernel(const bf16_t* __restrict__ A, const bf16_t* __restrict__ W,
                      const float* __restrict__ bias, const float* __restrict__ res,
                      float* __restrict__ outF, bf16_t* __restrict__ outB,
                      int M, int N, int K, float scaleOut) {
  constexpr int BM = 128, BN = 64, BK = 32, LDT = 40;  // 40 = 32 + 8 pad (16B aligned)
  __shared__ __attribute__((aligned(16))) bf16_t As[2][BM * LDT];
  __shared__ __attribute__((aligned(16))) bf16_t Bs[2][BN * LDT];

  const int tid  = threadIdx.x;
  const int lane = tid & 31;
  const int wave = tid >> 5;
  const int m0 = blockIdx.y * BM;
  const int n0 = blockIdx.x * BN;
  const int wm = (wave >> 1) * 32;   // 4 waves along M
  const int wn = (wave & 1) * 32;    // 2 waves along N

  // Per-thread staging slots (3 x b128 per thread per tile).
  const int eA0 = tid * 8,            rA0 = eA0 >> 5, cA0 = eA0 & 31;
  const int eA1 = (tid + 256) * 8,    rA1 = eA1 >> 5, cA1 = eA1 & 31;
  const int eB  = tid * 8,            rB  = eB  >> 5, cB  = eB  & 31;

  auto stage = [&](int buf, int k0) {
    async_copy_b128(&A[(size_t)(m0 + rA0) * K + k0 + cA0], &As[buf][rA0 * LDT + cA0]);
    async_copy_b128(&A[(size_t)(m0 + rA1) * K + k0 + cA1], &As[buf][rA1 * LDT + cA1]);
    async_copy_b128(&W[(size_t)(n0 + rB)  * K + k0 + cB],  &Bs[buf][rB  * LDT + cB]);
  };

  const v8f vzero = {0.f,0.f,0.f,0.f,0.f,0.f,0.f,0.f};
  v8f acc00 = vzero, acc01 = vzero, acc10 = vzero, acc11 = vzero;

  const int NT = K / BK;
  stage(0, 0);
  for (int t = 0; t < NT; ++t) {
    const int cur = t & 1;
    if (t + 1 < NT) {
      stage(cur ^ 1, (t + 1) * BK);
      asm volatile("s_wait_asynccnt 0x3" ::: "memory");  // tile t landed, t+1 in flight
    } else {
      asm volatile("s_wait_asynccnt 0x0" ::: "memory");
    }
    __syncthreads();

    v16bf a0 = frag_a_load(&As[cur][(wm +  0) * LDT], LDT, lane);
    v16bf a1 = frag_a_load(&As[cur][(wm + 16) * LDT], LDT, lane);
    v16bf b0 = frag_b_load(&Bs[cur][(wn +  0) * LDT], LDT, lane);
    v16bf b1 = frag_b_load(&Bs[cur][(wn + 16) * LDT], LDT, lane);

    acc00 = wmma_bf16(a0, b0, acc00);
    acc01 = wmma_bf16(a0, b1, acc01);
    acc10 = wmma_bf16(a1, b0, acc10);
    acc11 = wmma_bf16(a1, b1, acc11);

    __syncthreads();  // reads done before buf[cur] is overwritten next round
  }

  // C/D layout: element (m,n) -> vgpr r = m%8, lane = (n&15) + 16*(m/8 within 16-tile)
  v8f accs[4] = {acc00, acc01, acc10, acc11};
#pragma unroll
  for (int t = 0; t < 4; ++t) {
    const int mi = t >> 1, ni = t & 1;
#pragma unroll
    for (int r = 0; r < 8; ++r) {
      int m = m0 + wm + mi * 16 + r + ((lane & 16) ? 8 : 0);
      int n = n0 + wn + ni * 16 + (lane & 15);
      float v = accs[t][r];
      if constexpr (EPI == 0) {
        outB[(size_t)m * N + n] = __float2bfloat16((v + bias[n]) * scaleOut);
      } else if constexpr (EPI == 1) {
        v += bias[n];
        int b = m >> 10, s = m & (Ss - 1);
        int h = n >> 6,  dh = n & (DHd - 1);
        outB[((size_t)((b * Hh + h) * DHd + dh)) * Ss + s] = __float2bfloat16(v);
      } else if constexpr (EPI == 2) {
        outF[(size_t)m * N + n] = v + bias[n] + res[(size_t)m * N + n];
      } else {  // EPI == 3, exact GELU
        v += bias[n];
        v = 0.5f * v * (1.0f + erff(v * 0.70710678118f));
        outB[(size_t)m * N + n] = __float2bfloat16(v);
      }
    }
  }
}

// ---------------------------------------------------------------------------
// Fused flash-style attention, both branches of the reference:
//   dense : scores = QK^T*scale + (in_window ? 0 : -lam), softmax over all keys
//   sparse: scores = QK^T*scale masked to the window (-1e9 outside)
// selected per batch by lam = 10*exp(-5*u_prev[b]) >= 1.
// Q pre-scaled by 1/sqrt(dh) in its projection epilogue. V stored [B,H,DH,S].
// Block 256 threads = 8 independent waves; wave owns 16 query rows.
// ---------------------------------------------------------------------------
__global__ __launch_bounds__(256)
void flash_attn_kernel(const bf16_t* __restrict__ Qb, const bf16_t* __restrict__ Kb,
                       const bf16_t* __restrict__ Vt, const float* __restrict__ u_prev,
                       bf16_t* __restrict__ attnB) {
  __shared__ __attribute__((aligned(16))) bf16_t Pst[8][16 * 40];  // per-wave P staging

  const int tid  = threadIdx.x;
  const int lane = tid & 31;
  const int wave = tid >> 5;
  const int bh = blockIdx.y, b = bh >> 3, h = bh & 7;
  const int qbase = blockIdx.x * 128 + wave * 16;

  const float lam = 10.0f * __expf(-5.0f * u_prev[b]);
  const bool  sparse = (lam >= 1.0f);

  const bf16_t* qptr = Qb + ((size_t)(b * Ss + qbase)) * Dd + h * DHd;
  const v16bf qa0 = frag_a_load(qptr,      Dd, lane);  // d 0..31
  const v16bf qa1 = frag_a_load(qptr + 32, Dd, lane);  // d 32..63

  const v8f vzero = {0.f,0.f,0.f,0.f,0.f,0.f,0.f,0.f};
  v8f o0 = vzero, o1 = vzero, o2 = vzero, o3 = vzero;
  float mrow[8], lsum[8];
#pragma unroll
  for (int r = 0; r < 8; ++r) { mrow[r] = -3.0e38f; lsum[r] = 0.0f; }

  const int mrowoff = (lane & 16) ? 8 : 0;
  const int ncol = lane & 15;

  for (int kt = 0; kt < Ss; kt += 32) {
    // Sparse: wave-uniform skip of tiles fully outside the union window.
    if (sparse && (kt + 31 < qbase - WHalf || kt > qbase + 15 + WHalf)) continue;

    const bf16_t* kptr = Kb + ((size_t)(b * Ss + kt)) * Dd + h * DHd;
    v16bf kb00 = frag_b_load(kptr,                Dd, lane);  // keys 0-15,  d 0-31
    v16bf kb01 = frag_b_load(kptr + 32,           Dd, lane);  // keys 0-15,  d 32-63
    v16bf kb10 = frag_b_load(kptr + 16 * Dd,      Dd, lane);  // keys 16-31, d 0-31
    v16bf kb11 = frag_b_load(kptr + 16 * Dd + 32, Dd, lane);  // keys 16-31, d 32-63

    v8f s0 = wmma_bf16(qa0, kb00, vzero); s0 = wmma_bf16(qa1, kb01, s0);
    v8f s1 = wmma_bf16(qa0, kb10, vzero); s1 = wmma_bf16(qa1, kb11, s1);

#pragma unroll
    for (int r = 0; r < 8; ++r) {
      const int q  = qbase + r + mrowoff;
      const int j0 = kt + ncol, j1 = j0 + 16;
      float e0 = s0[r], e1 = s1[r];
      const bool w0 = (j0 >= q - WHalf) && (j0 <= q + WHalf);
      const bool w1 = (j1 >= q - WHalf) && (j1 <= q + WHalf);
      if (sparse) { e0 = w0 ? e0 : NEGV;  e1 = w1 ? e1 : NEGV; }
      else        { e0 += w0 ? 0.f : -lam; e1 += w1 ? 0.f : -lam; }

      // per-row max / sum across the 16-lane half, all in VALU via DPP16
      const float tmax = rowmax16(fmaxf(e0, e1));
      const float mnew = fmaxf(mrow[r], tmax);
      const float sf = __expf(mrow[r] - mnew);
      const float p0 = __expf(e0 - mnew);
      const float p1 = __expf(e1 - mnew);
      const float ps = rowsum16(p0 + p1);
      lsum[r] = lsum[r] * sf + ps;
      mrow[r] = mnew;
      o0[r] *= sf; o1[r] *= sf; o2[r] *= sf; o3[r] *= sf;

      // C-layout -> A-layout transpose of P through wave-private LDS
      const int prow = r + mrowoff;
      Pst[wave][prow * 40 + ncol]      = __float2bfloat16(p0);
      Pst[wave][prow * 40 + 16 + ncol] = __float2bfloat16(p1);
    }

    // Wave-local LDS RAW fence (no block barrier needed: staging is per-wave)
    asm volatile("s_wait_dscnt 0" ::: "memory");
    v16bf pa = frag_a_load(&Pst[wave][0], 40, lane);

    const bf16_t* vptr = Vt + ((size_t)bh * DHd) * Ss + kt;
    o0 = wmma_bf16(pa, frag_b_load(vptr + (size_t)0  * Ss, Ss, lane), o0);
    o1 = wmma_bf16(pa, frag_b_load(vptr + (size_t)16 * Ss, Ss, lane), o1);
    o2 = wmma_bf16(pa, frag_b_load(vptr + (size_t)32 * Ss, Ss, lane), o2);
    o3 = wmma_bf16(pa, frag_b_load(vptr + (size_t)48 * Ss, Ss, lane), o3);
  }

#pragma unroll
  for (int r = 0; r < 8; ++r) {
    const float inv = 1.0f / lsum[r];
    const int m = qbase + r + mrowoff;
    const size_t rowoff = ((size_t)(b * Ss + m)) * Dd + h * DHd + ncol;
    attnB[rowoff +  0] = __float2bfloat16(o0[r] * inv);
    attnB[rowoff + 16] = __float2bfloat16(o1[r] * inv);
    attnB[rowoff + 32] = __float2bfloat16(o2[r] * inv);
    attnB[rowoff + 48] = __float2bfloat16(o3[r] * inv);
  }
}

// --------------------------- small helper kernels ---------------------------

__global__ void cvt_bf16_kernel(const float* __restrict__ src, bf16_t* __restrict__ dst, int n) {
  int i = blockIdx.x * blockDim.x + threadIdx.x;
  if (i < n) dst[i] = __float2bfloat16(src[i]);
}

__global__ __launch_bounds__(128)
void ln_bf16_kernel(const float* __restrict__ in, const float* __restrict__ g,
                    const float* __restrict__ be, bf16_t* __restrict__ outB) {
  const int row = blockIdx.x;
  const float* x = in + (size_t)row * Dd;
  float v[4], s = 0.f, s2 = 0.f;
#pragma unroll
  for (int i = 0; i < 4; ++i) { v[i] = x[threadIdx.x + 128 * i]; s += v[i]; s2 += v[i] * v[i]; }
  __shared__ float rs[128], rs2[128];
  rs[threadIdx.x] = s; rs2[threadIdx.x] = s2; __syncthreads();
  for (int off = 64; off > 0; off >>= 1) {
    if (threadIdx.x < off) { rs[threadIdx.x] += rs[threadIdx.x + off]; rs2[threadIdx.x] += rs2[threadIdx.x + off]; }
    __syncthreads();
  }
  const float mean = rs[0] * (1.0f / Dd);
  const float var  = rs2[0] * (1.0f / Dd) - mean * mean;
  const float rstd = rsqrtf(var + 1e-5f);
#pragma unroll
  for (int i = 0; i < 4; ++i) {
    int c = threadIdx.x + 128 * i;
    outB[(size_t)row * Dd + c] = __float2bfloat16((v[i] - mean) * rstd * g[c] + be[c]);
  }
}

__global__ void zero_agg_kernel(float* agg) { if ((int)threadIdx.x < Bb) agg[threadIdx.x] = 0.f; }

__global__ __launch_bounds__(128)
void u_kernel(const float* __restrict__ out, const float* __restrict__ x, float* __restrict__ agg) {
  const int row = blockIdx.x;
  const float* o  = out + (size_t)row * Dd;
  const float* xr = x   + (size_t)row * Dd;
  float d2 = 0.f, x2 = 0.f;
#pragma unroll
  for (int i = 0; i < 4; ++i) {
    int c = threadIdx.x + 128 * i;
    float dv = o[c] - xr[c]; d2 += dv * dv;
    float xv = xr[c];        x2 += xv * xv;
  }
  __shared__ float rd[128], rx[128];
  rd[threadIdx.x] = d2; rx[threadIdx.x] = x2; __syncthreads();
  for (int off = 64; off > 0; off >>= 1) {
    if (threadIdx.x < off) { rd[threadIdx.x] += rd[threadIdx.x + off]; rx[threadIdx.x] += rx[threadIdx.x + off]; }
    __syncthreads();
  }
  if (threadIdx.x == 0) {
    float u = sqrtf(rd[0]) / (sqrtf(rx[0]) + 1e-8f);
    atomicAdd(&agg[row >> 10], u * (1.0f / (float)Ss));
  }
}

// --------------------------------- launch -----------------------------------

extern "C" void kernel_launch(void* const* d_in, const int* in_sizes, int n_in,
                              void* d_out, int out_size, void* d_ws, size_t ws_size,
                              hipStream_t stream) {
  (void)in_sizes; (void)n_in; (void)out_size; (void)ws_size;

  const float* x    = (const float*)d_in[0];
  const float* upv  = (const float*)d_in[1];
  const float* wq   = (const float*)d_in[2];  const float* bq = (const float*)d_in[3];
  const float* wk   = (const float*)d_in[4];  const float* bk = (const float*)d_in[5];
  const float* wv   = (const float*)d_in[6];  const float* bv = (const float*)d_in[7];
  const float* wo   = (const float*)d_in[8];  const float* bo = (const float*)d_in[9];
  const float* ln1g = (const float*)d_in[10]; const float* ln1b = (const float*)d_in[11];
  const float* ln2g = (const float*)d_in[12]; const float* ln2b = (const float*)d_in[13];
  const float* w1   = (const float*)d_in[14]; const float* b1 = (const float*)d_in[15];
  const float* w2   = (const float*)d_in[16]; const float* b2 = (const float*)d_in[17];
  // d_in[18..20] (prior_mask / indices / index_mask) are recomputed arithmetically.

  char* ws = (char*)d_ws;
  size_t off = 0;
  auto alloc = [&](size_t bytes) -> void* {
    void* p = ws + off; off += (bytes + 255) & ~(size_t)255; return p;
  };
  const size_t MT = (size_t)Bb * Ss;  // 2048 tokens

  bf16_t* xn_b = (bf16_t*)alloc(MT * Dd * 2);
  bf16_t* q_b  = (bf16_t*)alloc(MT * Dd * 2);
  bf16_t* k_b  = (bf16_t*)alloc(MT * Dd * 2);
  bf16_t* vt_b = (bf16_t*)alloc(MT * Dd * 2);   // [B,H,DH,S]
  bf16_t* at_b = (bf16_t*)alloc(MT * Dd * 2);
  bf16_t* hn_b = (bf16_t*)alloc(MT * Dd * 2);
  bf16_t* g_b  = (bf16_t*)alloc(MT * (size_t)Ff * 2);
  float*  h_f  = (float*) alloc(MT * Dd * 4);
  bf16_t* wq_b = (bf16_t*)alloc((size_t)Dd * Dd * 2);
  bf16_t* wk_b = (bf16_t*)alloc((size_t)Dd * Dd * 2);
  bf16_t* wv_b = (bf16_t*)alloc((size_t)Dd * Dd * 2);
  bf16_t* wo_b = (bf16_t*)alloc((size_t)Dd * Dd * 2);
  bf16_t* w1_b = (bf16_t*)alloc((size_t)Ff * Dd * 2);
  bf16_t* w2_b = (bf16_t*)alloc((size_t)Dd * Ff * 2);

  auto cvt = [&](const float* s, bf16_t* d, int n) {
    cvt_bf16_kernel<<<(n + 255) / 256, 256, 0, stream>>>(s, d, n);
  };
  cvt(wq, wq_b, Dd * Dd); cvt(wk, wk_b, Dd * Dd);
  cvt(wv, wv_b, Dd * Dd); cvt(wo, wo_b, Dd * Dd);
  cvt(w1, w1_b, Ff * Dd); cvt(w2, w2_b, Dd * Ff);

  ln_bf16_kernel<<<(unsigned)MT, 128, 0, stream>>>(x, ln1g, ln1b, xn_b);

  const dim3 blk(256);
  const dim3 gD(Dd / 64, (unsigned)(MT / 128));   // N=512 tiles
  const dim3 gF(Ff / 64, (unsigned)(MT / 128));   // N=2048 tiles

  // Q pre-scaled by 1/sqrt(dh)=0.125 (bias included, matching reference order)
  gemm_bf16_kernel<0><<<gD, blk, 0, stream>>>(xn_b, wq_b, bq, nullptr, nullptr, q_b,
                                              (int)MT, Dd, Dd, 0.125f);
  gemm_bf16_kernel<0><<<gD, blk, 0, stream>>>(xn_b, wk_b, bk, nullptr, nullptr, k_b,
                                              (int)MT, Dd, Dd, 1.0f);
  gemm_bf16_kernel<1><<<gD, blk, 0, stream>>>(xn_b, wv_b, bv, nullptr, nullptr, vt_b,
                                              (int)MT, Dd, Dd, 1.0f);

  flash_attn_kernel<<<dim3(Ss / 128, Bb * Hh), blk, 0, stream>>>(q_b, k_b, vt_b, upv, at_b);

  // h = x + attn @ wo^T + bo
  gemm_bf16_kernel<2><<<gD, blk, 0, stream>>>(at_b, wo_b, bo, x, h_f, nullptr,
                                              (int)MT, Dd, Dd, 1.0f);

  ln_bf16_kernel<<<(unsigned)MT, 128, 0, stream>>>(h_f, ln2g, ln2b, hn_b);

  // g = gelu(hn @ w1^T + b1)
  gemm_bf16_kernel<3><<<gF, blk, 0, stream>>>(hn_b, w1_b, b1, nullptr, nullptr, g_b,
                                              (int)MT, Ff, Dd, 1.0f);

  // out = h + g @ w2^T + b2  (written straight to d_out)
  float* outp = (float*)d_out;
  gemm_bf16_kernel<2><<<gD, blk, 0, stream>>>(g_b, w2_b, b2, h_f, outp, nullptr,
                                              (int)MT, Dd, Ff, 1.0f);

  // aggregated_u appended after out
  float* agg = outp + MT * Dd;
  zero_agg_kernel<<<1, 32, 0, stream>>>(agg);
  u_kernel<<<(unsigned)MT, 128, 0, stream>>>(outp, x, agg);
}
